// DisentangledAttn_87247965651111
// MI455X (gfx1250) — matrix operand order repaired and checked by reading, not compiled, not run on hardware
//
#include <hip/hip_runtime.h>
#include <hip/hip_bf16.h>
#include <math.h>

#define Bb   4
#define Nn   1024
#define Dd   1024
#define Hh   8
#define Ll   512
#define DK   128

typedef __attribute__((ext_vector_type(16))) __bf16 v16bf;
typedef __attribute__((ext_vector_type(8)))  __bf16 v8bf;
typedef __attribute__((ext_vector_type(8)))  float  v8f;

// ---------------------------------------------------------------------------
// WMMA fragment loaders (CDNA5 wave32 layouts, CDNA5 ISA §7.12.2)
// A 16x32 bf16: lane L holds row M=L&15; vgpr j=0..3 -> K pairs {0..7|8..15},
// j=4..7 -> K {16..23|24..31} depending on lane half. Two 16B loads.
// ---------------------------------------------------------------------------
__device__ __forceinline__ v16bf load_a_frag(const __bf16* tile, int ld, int k0, int lane) {
  const __bf16* row = tile + (long)(lane & 15) * ld + k0;
  int off = (lane & 16) ? 8 : 0;
  v8bf lo = *(const v8bf*)(row + off);
  v8bf hi = *(const v8bf*)(row + 16 + off);
  return __builtin_shufflevector(lo, hi, 0,1,2,3,4,5,6,7,8,9,10,11,12,13,14,15);
}

// B 32x16 bf16 where the source is row-major [col][k] (i.e. B = src^T):
// lane holds col=lane&15, K same order as A-frag: {0..7|8..15}, {16..23|24..31} -> two 16B loads.
__device__ __forceinline__ v16bf load_bt_frag(const __bf16* tile, int ld, int k0, int lane) {
  const __bf16* p = tile + (long)(lane & 15) * ld + k0 + ((lane & 16) ? 8 : 0);
  v8bf lo = *(const v8bf*)p;
  v8bf hi = *(const v8bf*)(p + 16);
  return __builtin_shufflevector(lo, hi, 0,1,2,3,4,5,6,7,8,9,10,11,12,13,14,15);
}

__device__ __forceinline__ v8f wmma_bf16(v16bf a, v16bf b, v8f c) {
  return __builtin_amdgcn_wmma_f32_16x16x32_bf16(false, a, false, b, (short)0, c, false, false);
}

// ---------------------------------------------------------------------------
// Conversions
// ---------------------------------------------------------------------------
__global__ void k_cvt_bf16(const float* __restrict__ in, __bf16* __restrict__ out, int n) {
  int i = blockIdx.x * 256 + threadIdx.x;
  if (i < n) out[i] = (__bf16)in[i];
}

// in [rows][cols] fp32 -> out [cols][rows] bf16 (weight transpose for B-frags)
__global__ void k_cvt_t_bf16(const float* __restrict__ in, __bf16* __restrict__ out,
                             int rows, int cols) {
  long i = (long)blockIdx.x * 256 + threadIdx.x;
  long n = (long)rows * cols;
  if (i < n) {
    int r = (int)(i / cols), c = (int)(i % cols);
    out[(long)c * rows + r] = (__bf16)in[i];
  }
}

// ---------------------------------------------------------------------------
// Generic projection GEMM: C[M,Nc] = A[M,K] @ Wt^T + bias, all WMMA bf16.
// 128 threads = 4 waves; block tile 16 rows x 256 cols; each wave owns 4
// column tiles and reuses one A-fragment per K-step across them
// (≈2.5 VMEM loads per WMMA instead of 4).
// mode 0: head-major bf16   out[((b*H+h)*N+n)*DK+d]        (q/k, scale folded)
// mode 2: v transposed bf16 out[((b*H+h)*DK+d)*N+n]
// mode 3: rel-proj bf16     out[((hoff+c/DK)*L+m)*DK+d]
// mode 4: fp32 row-major    out[m*Nc+c]                    (final projection)
// ---------------------------------------------------------------------------
__global__ void __launch_bounds__(128) k_proj_gemm(
    const __bf16* __restrict__ A, const __bf16* __restrict__ Wt,
    const float* __restrict__ bias, int Kdim, int Nc,
    int mode, int hoff, float scl,
    __bf16* __restrict__ outB, float* __restrict__ outF) {
  int lane = threadIdx.x & 31;
  int w    = threadIdx.x >> 5;
  int m0   = blockIdx.x * 16;
  int c0   = blockIdx.y * 256 + w * 64;
  const __bf16* At = A + (long)m0 * Kdim;
  v8f acc[4] = {};
  for (int k0 = 0; k0 < Kdim; k0 += 32) {
    v16bf a = load_a_frag(At, Kdim, k0, lane);
#pragma unroll
    for (int t = 0; t < 4; ++t)
      acc[t] = wmma_bf16(a, load_bt_frag(Wt + (long)(c0 + t * 16) * Kdim, Kdim, k0, lane),
                         acc[t]);
  }
#pragma unroll
  for (int t = 0; t < 4; ++t) {
    int c = c0 + t * 16 + (lane & 15);
    float bv = bias[c];
#pragma unroll
    for (int r = 0; r < 8; ++r) {
      int row = ((lane & 16) ? 8 : 0) + r;
      int m   = m0 + row;
      float val = (acc[t][r] + bv) * scl;
      if (mode == 0) {
        int b_ = m >> 10, n = m & (Nn - 1), h = c >> 7, d = c & (DK - 1);
        outB[((long)(b_ * Hh + h) * Nn + n) * DK + d] = (__bf16)val;
      } else if (mode == 2) {
        int b_ = m >> 10, n = m & (Nn - 1), h = c >> 7, d = c & (DK - 1);
        outB[((long)(b_ * Hh + h) * DK + d) * Nn + n] = (__bf16)val;
      } else if (mode == 3) {
        int h = hoff + (c >> 7), d = c & (DK - 1);
        outB[((long)h * Ll + m) * DK + d] = (__bf16)val;
      } else {
        outF[(long)m * Nc + c] = val;
      }
    }
  }
}

// ---------------------------------------------------------------------------
// Table GEMM: Out[z][m][n] = A[z][m,:DK] . B[z][n,:DK]  (both bf16 row-major)
// z = b*H + h ; per-batch/head strides passed separately. Same 16x256 block
// tile with A-fragment reuse.
// Ptab = lq_scaled . k^T   [L,N] ;  Ctab = q_scaled . lk^T  [N,L]
// ---------------------------------------------------------------------------
__global__ void __launch_bounds__(128) k_tab_gemm(
    const __bf16* __restrict__ Abase, long aSb, long aSh,
    const __bf16* __restrict__ Bbase, long bSb, long bSh,
    float* __restrict__ Out, int M2, int N2) {
  int lane = threadIdx.x & 31;
  int w    = threadIdx.x >> 5;
  int m0   = blockIdx.x * 16;
  int c0   = blockIdx.y * 256 + w * 64;
  int z = blockIdx.z, b_ = z >> 3, h = z & 7;
  const __bf16* At = Abase + b_ * aSb + h * aSh + (long)m0 * DK;
  const __bf16* Bt = Bbase + b_ * bSb + h * bSh;
  v8f acc[4] = {};
#pragma unroll
  for (int k0 = 0; k0 < DK; k0 += 32) {
    v16bf a = load_a_frag(At, DK, k0, lane);
#pragma unroll
    for (int t = 0; t < 4; ++t)
      acc[t] = wmma_bf16(a, load_bt_frag(Bt + (long)(c0 + t * 16) * DK, DK, k0, lane),
                         acc[t]);
  }
  float* Oz = Out + (long)z * M2 * N2;
#pragma unroll
  for (int t = 0; t < 4; ++t) {
    int c = c0 + t * 16 + (lane & 15);
#pragma unroll
    for (int r = 0; r < 8; ++r) {
      int row = ((lane & 16) ? 8 : 0) + r;
      Oz[(long)(m0 + row) * N2 + c] = acc[t][r];
    }
  }
}

// ---------------------------------------------------------------------------
// Fused attention: block = (i-tile of 16 rows, h, b); 4 waves.
// Flash-style over 4 key-chunks of 256: WMMA c2c (q-fragments hoisted into
// registers for the whole block) -> gather p2c/c2p -> mask -> parallel online
// softmax (8 threads/row, LDS tree reduce) -> WMMA probs @ v^T with running
// rescale. Scores never touch HBM.
// ---------------------------------------------------------------------------
__global__ void __launch_bounds__(128) k_attn(
    const __bf16* __restrict__ qbf, const __bf16* __restrict__ kbf,
    const __bf16* __restrict__ vT,
    const float* __restrict__ Ptab, const float* __restrict__ Ctab,
    const int* __restrict__ rel, const int* __restrict__ mask,
    __bf16* __restrict__ attout) {
  __shared__ __attribute__((aligned(32))) float  sc[16][256];
  __shared__ __attribute__((aligned(32))) __bf16 probs[16][256];
  __shared__ float red[16][8];
  __shared__ float rowm[16], rowsum[16], rowf[16];

  int lane = threadIdx.x & 31;
  int w    = threadIdx.x >> 5;
  int tid  = threadIdx.x;
  int i0 = blockIdx.x * 16;
  int h  = blockIdx.y;
  int b_ = blockIdx.z;
  long bh = (long)b_ * Hh + h;
  const __bf16* qh = qbf + bh * Nn * DK;
  const __bf16* kh = kbf + bh * Nn * DK;
  const __bf16* vh = vT  + bh * DK * Nn;
  const float*  Ph = Ptab + bh * Ll * Nn;
  const float*  Ch = Ctab + bh * Nn * Ll;
  const int* relh  = rel  + bh * (long)Nn * Nn;
  const int* maskb = mask + (long)b_ * Nn * Nn;

  // q fragments for all 4 K-steps: invariant for the whole block.
  v16bf qfrag[4];
#pragma unroll
  for (int kk = 0; kk < 4; ++kk)
    qfrag[kk] = load_a_frag(qh + (long)i0 * DK, DK, kk * 32, lane);

  if (tid < 16) { rowm[tid] = -3.0e38f; rowsum[tid] = 0.f; }
  v8f acc0 = {}, acc1 = {};
  int srow = tid >> 3, spart = tid & 7;          // softmax: 8 threads per row
  __syncthreads();

  for (int chunk = 0; chunk < 4; ++chunk) {
    int j0 = chunk * 256;
    // ---- score tiles: wave w owns column tiles 4w..4w+3 of this chunk ----
    for (int t = 0; t < 4; ++t) {
      int jt = j0 + (w * 4 + t) * 16;
      v8f s = {};
#pragma unroll
      for (int kk = 0; kk < 4; ++kk)
        s = wmma_bf16(qfrag[kk], load_bt_frag(kh + (long)jt * DK, DK, kk * 32, lane), s);
      int col = lane & 15;
      int j = jt + col;
#pragma unroll
      for (int r = 0; r < 8; ++r) {
        int irow = ((lane & 16) ? 8 : 0) + r;
        int i = i0 + irow;
        int rij = relh[(long)i * Nn + j];   // c2p gather index
        int rji = relh[(long)j * Nn + i];   // p2c gather index (transposed)
        float val = s[r] + Ch[(long)i * Ll + rij] + Ph[(long)rji * Nn + j];
        if (maskb[(long)i * Nn + j] == 1) val = -1e9f;
        sc[irow][(jt - j0) + col] = val;
      }
    }
    __syncthreads();
    // ---- parallel online softmax: partial max ----
    {
      float pm = -3.0e38f;
      int jb = spart * 32;
      for (int jj = jb; jj < jb + 32; ++jj) pm = fmaxf(pm, sc[srow][jj]);
      red[srow][spart] = pm;
    }
    __syncthreads();
    if (tid < 16) {
      float rmax = red[tid][0];
#pragma unroll
      for (int u = 1; u < 8; ++u) rmax = fmaxf(rmax, red[tid][u]);
      float nm = fmaxf(rowm[tid], rmax);
      rowf[tid] = __expf(rowm[tid] - nm);
      rowm[tid] = nm;
    }
    __syncthreads();
    // ---- exp + partial sum, probs to LDS as bf16 ----
    {
      float nm = rowm[srow], psum = 0.f;
      int jb = spart * 32;
      for (int jj = jb; jj < jb + 32; ++jj) {
        float pv = __expf(sc[srow][jj] - nm);
        probs[srow][jj] = (__bf16)pv;
        psum += pv;
      }
      red[srow][spart] = psum;
    }
    __syncthreads();
    if (tid < 16) {
      float sum = red[tid][0];
#pragma unroll
      for (int u = 1; u < 8; ++u) sum += red[tid][u];
      rowsum[tid] = rowsum[tid] * rowf[tid] + sum;
    }
    __syncthreads();
    // ---- rescale accumulators, then P @ V^T (wave w owns d-cols [32w,32w+32)) ----
#pragma unroll
    for (int r = 0; r < 8; ++r) {
      int irow = ((lane & 16) ? 8 : 0) + r;
      float f = rowf[irow];
      acc0[r] *= f; acc1[r] *= f;
    }
#pragma unroll
    for (int k0 = 0; k0 < 256; k0 += 32) {
      v16bf a = load_a_frag(&probs[0][0], 256, k0, lane);
      acc0 = wmma_bf16(a, load_bt_frag(vh + (long)(w * 32     ) * Nn + j0, Nn, k0, lane), acc0);
      acc1 = wmma_bf16(a, load_bt_frag(vh + (long)(w * 32 + 16) * Nn + j0, Nn, k0, lane), acc1);
    }
    __syncthreads();
  }
  // ---- finalize: attout bf16 in [B,N,D] layout for the output projection ----
  int col = lane & 15;
#pragma unroll
  for (int r = 0; r < 8; ++r) {
    int irow = ((lane & 16) ? 8 : 0) + r;
    int i = i0 + irow;
    float inv = 1.0f / rowsum[irow];
    long base = ((long)(b_ * Nn + i)) * Dd + h * DK + w * 32;
    attout[base + col]      = (__bf16)(acc0[r] * inv);
    attout[base + 16 + col] = (__bf16)(acc1[r] * inv);
  }
}

// ---------------------------------------------------------------------------
extern "C" void kernel_launch(void* const* d_in, const int* in_sizes, int n_in,
                              void* d_out, int out_size, void* d_ws, size_t ws_size,
                              hipStream_t stream) {
  (void)in_sizes; (void)n_in; (void)out_size; (void)ws_size;
  const float* query   = (const float*)d_in[0];
  const float* key_    = (const float*)d_in[1];
  const float* value   = (const float*)d_in[2];
  const float* rel_emb = (const float*)d_in[3];
  const int*   rel     = (const int*)d_in[4];
  const int*   mask    = (const int*)d_in[5];
  const float* Wq = (const float*)d_in[6];  const float* bq = (const float*)d_in[7];
  const float* Wk = (const float*)d_in[8];  const float* bk = (const float*)d_in[9];
  const float* Wv = (const float*)d_in[10]; const float* bv = (const float*)d_in[11];
  const float* Wo = (const float*)d_in[12]; const float* bo = (const float*)d_in[13];
  const float* Wlq = (const float*)d_in[14]; const float* blq = (const float*)d_in[15];
  const float* Wlk = (const float*)d_in[16]; const float* blk = (const float*)d_in[17];
  const float* Wtq = (const float*)d_in[18]; const float* btq = (const float*)d_in[19];
  const float* Wtk = (const float*)d_in[20]; const float* btk = (const float*)d_in[21];

  char* p = (char*)d_ws;
  auto alloc = [&](size_t bytes) { char* r = p; p += (bytes + 255) & ~(size_t)255; return r; };
  const long MD = (long)Bb * Nn;               // 4096
  __bf16* xq   = (__bf16*)alloc(MD * Dd * 2);
  __bf16* xk   = (__bf16*)alloc(MD * Dd * 2);
  __bf16* xv   = (__bf16*)alloc(MD * Dd * 2);
  __bf16* WqT  = (__bf16*)alloc((long)Dd * Dd * 2);
  __bf16* WkT  = (__bf16*)alloc((long)Dd * Dd * 2);
  __bf16* WvT  = (__bf16*)alloc((long)Dd * Dd * 2);
  __bf16* WoT  = (__bf16*)alloc((long)Dd * Dd * 2);
  __bf16* WlqT = (__bf16*)alloc((long)512 * Dd * 2);
  __bf16* WlkT = (__bf16*)alloc((long)512 * Dd * 2);
  __bf16* WtqT = (__bf16*)alloc((long)512 * Dd * 2);
  __bf16* WtkT = (__bf16*)alloc((long)512 * Dd * 2);
  __bf16* lbf  = (__bf16*)alloc((long)Ll * Dd * 2);
  __bf16* tbf  = (__bf16*)alloc((long)Ll * Dd * 2);
  __bf16* qbf  = (__bf16*)alloc((long)Bb * Hh * Nn * DK * 2);
  __bf16* kbf  = (__bf16*)alloc((long)Bb * Hh * Nn * DK * 2);
  __bf16* vTb  = (__bf16*)alloc((long)Bb * Hh * DK * Nn * 2);
  __bf16* lqbf = (__bf16*)alloc((long)Hh * Ll * DK * 2);
  __bf16* lkbf = (__bf16*)alloc((long)Hh * Ll * DK * 2);
  float*  Ptab = (float*)alloc((long)Bb * Hh * Ll * Nn * 4);
  float*  Ctab = (float*)alloc((long)Bb * Hh * Nn * Ll * 4);
  __bf16* atto = (__bf16*)alloc(MD * Dd * 2);

  const int nBig = (int)(MD * Dd);             // 4,194,304
  k_cvt_bf16<<<nBig / 256, 256, 0, stream>>>(query, xq, nBig);
  k_cvt_bf16<<<nBig / 256, 256, 0, stream>>>(key_,  xk, nBig);
  k_cvt_bf16<<<nBig / 256, 256, 0, stream>>>(value, xv, nBig);
  const int nRel = Ll * Dd;                    // 524,288
  k_cvt_bf16<<<nRel / 256, 256, 0, stream>>>(rel_emb,        lbf, nRel);
  k_cvt_bf16<<<nRel / 256, 256, 0, stream>>>(rel_emb + nRel, tbf, nRel);

  k_cvt_t_bf16<<<(Dd * Dd) / 256, 256, 0, stream>>>(Wq, WqT, Dd, Dd);
  k_cvt_t_bf16<<<(Dd * Dd) / 256, 256, 0, stream>>>(Wk, WkT, Dd, Dd);
  k_cvt_t_bf16<<<(Dd * Dd) / 256, 256, 0, stream>>>(Wv, WvT, Dd, Dd);
  k_cvt_t_bf16<<<(Dd * Dd) / 256, 256, 0, stream>>>(Wo, WoT, Dd, Dd);
  k_cvt_t_bf16<<<(Dd * 512) / 256, 256, 0, stream>>>(Wlq, WlqT, Dd, 512);
  k_cvt_t_bf16<<<(Dd * 512) / 256, 256, 0, stream>>>(Wlk, WlkT, Dd, 512);
  k_cvt_t_bf16<<<(Dd * 512) / 256, 256, 0, stream>>>(Wtq, WtqT, Dd, 512);
  k_cvt_t_bf16<<<(Dd * 512) / 256, 256, 0, stream>>>(Wtk, WtkT, Dd, 512);

  const float rs = 1.0f / sqrtf(3.0f * (float)DK);   // fold 1/sqrt(3*dk) into q, lq
  dim3 gqkv(MD / 16, Dd / 256);                      // (256, 4)
  k_proj_gemm<<<gqkv, 128, 0, stream>>>(xq, WqT, bq, Dd, Dd, 0, 0, rs,   qbf, nullptr);
  k_proj_gemm<<<gqkv, 128, 0, stream>>>(xk, WkT, bk, Dd, Dd, 0, 0, 1.0f, kbf, nullptr);
  k_proj_gemm<<<gqkv, 128, 0, stream>>>(xv, WvT, bv, Dd, Dd, 2, 0, 1.0f, vTb, nullptr);
  dim3 grel(Ll / 16, 512 / 256);                     // (32, 2)
  k_proj_gemm<<<grel, 128, 0, stream>>>(lbf, WlqT, blq, Dd, 512, 3, 0, rs,   lqbf, nullptr);
  k_proj_gemm<<<grel, 128, 0, stream>>>(tbf, WtqT, btq, Dd, 512, 3, 4, rs,   lqbf, nullptr);
  k_proj_gemm<<<grel, 128, 0, stream>>>(lbf, WlkT, blk, Dd, 512, 3, 0, 1.0f, lkbf, nullptr);
  k_proj_gemm<<<grel, 128, 0, stream>>>(tbf, WtkT, btk, Dd, 512, 3, 4, 1.0f, lkbf, nullptr);

  // Ptab[bh][l][n] = lq . k^T ; Ctab[bh][i][l] = q . lk^T
  dim3 gp(Ll / 16, Nn / 256, Bb * Hh);               // (32, 4, 32)
  k_tab_gemm<<<gp, 128, 0, stream>>>(lqbf, 0L, (long)Ll * DK,
                                     kbf, (long)Hh * Nn * DK, (long)Nn * DK,
                                     Ptab, Ll, Nn);
  dim3 gc(Nn / 16, Ll / 256, Bb * Hh);               // (64, 2, 32)
  k_tab_gemm<<<gc, 128, 0, stream>>>(qbf, (long)Hh * Nn * DK, (long)Nn * DK,
                                     lkbf, 0L, (long)Ll * DK,
                                     Ctab, Nn, Ll);

  dim3 ga(Nn / 16, Hh, Bb);
  k_attn<<<ga, 128, 0, stream>>>(qbf, kbf, vTb, Ptab, Ctab, rel, mask, atto);

  k_proj_gemm<<<gqkv, 128, 0, stream>>>(atto, WoT, bo, Dd, Dd, 4, 0, 1.0f,
                                        nullptr, (float*)d_out);
}
